// TherapeuticMoEModel_49435073577790
// MI455X (gfx1250) — matrix operand order repaired and checked by reading, not compiled, not run on hardware
//
#include <hip/hip_runtime.h>

// ---------------------------------------------------------------------------
// Problem constants (match reference)
// ---------------------------------------------------------------------------
#define LL 4
#define EE 4
#define HH 1024
#define FF 4096
#define BB 4
#define SS 2048
#define NTOK (BB * SS)   // 8192 tokens

typedef __attribute__((ext_vector_type(16))) __bf16 v16bf;
typedef __attribute__((ext_vector_type(8)))  float  v8f;

__device__ __forceinline__ unsigned short f2bf(float f) {
    unsigned int u = __float_as_uint(f);
    u += 0x7FFFu + ((u >> 16) & 1u);            // round-to-nearest-even
    return (unsigned short)(u >> 16);
}

__device__ __forceinline__ v8f zero8() {
    v8f z;
#pragma unroll
    for (int i = 0; i < 8; ++i) z[i] = 0.0f;
    return z;
}

// ---------------------------------------------------------------------------
// fp32 -> bf16 conversion (vectorized by 4)
// ---------------------------------------------------------------------------
__global__ void cvt_f32_bf16(const float* __restrict__ in,
                             unsigned short* __restrict__ out, int n4) {
    int i = blockIdx.x * blockDim.x + threadIdx.x;
    if (i >= n4) return;
    float4 v = ((const float4*)in)[i];
    uint2 p;
    p.x = (unsigned)f2bf(v.x) | ((unsigned)f2bf(v.y) << 16);
    p.y = (unsigned)f2bf(v.z) | ((unsigned)f2bf(v.w) << 16);
    ((uint2*)out)[i] = p;
}

// ---------------------------------------------------------------------------
// residual init: acc = x  (vectorized by 4)
// ---------------------------------------------------------------------------
__global__ void copy_f32(const float* __restrict__ in, float* __restrict__ out, int n4) {
    int i = blockIdx.x * blockDim.x + threadIdx.x;
    if (i >= n4) return;
    ((float4*)out)[i] = ((const float4*)in)[i];
}

// ---------------------------------------------------------------------------
// xe = bf16(xhat * g + b)   (per expert; vectorized by 4, H/4 = 256 pow2)
// ---------------------------------------------------------------------------
__global__ void apply_gb(const float* __restrict__ xhat, const float* __restrict__ g,
                         const float* __restrict__ b, unsigned short* __restrict__ xe,
                         int total4) {
    int i = blockIdx.x * blockDim.x + threadIdx.x;
    if (i >= total4) return;
    int h4 = i & (HH / 4 - 1);
    float4 xv = ((const float4*)xhat)[i];
    float4 gv = ((const float4*)g)[h4];
    float4 bv = ((const float4*)b)[h4];
    uint2 p;
    p.x = (unsigned)f2bf(fmaf(xv.x, gv.x, bv.x)) |
          ((unsigned)f2bf(fmaf(xv.y, gv.y, bv.y)) << 16);
    p.y = (unsigned)f2bf(fmaf(xv.z, gv.z, bv.z)) |
          ((unsigned)f2bf(fmaf(xv.w, gv.w, bv.w)) << 16);
    ((uint2*)xe)[i] = p;
}

// ---------------------------------------------------------------------------
// Router + shared LayerNorm stats.  One block (256 thr) per token.
// Writes xhat (fp32, [NTOK,H]) and renormalized top-2 weights wt ([NTOK,E]).
// ---------------------------------------------------------------------------
__global__ __launch_bounds__(256) void router_ln(const float* __restrict__ x,
                                                 const float* __restrict__ rw,
                                                 const float* __restrict__ rb,
                                                 float* __restrict__ xhat,
                                                 float* __restrict__ wt) {
    const int tok = blockIdx.x;
    const int tid = threadIdx.x;
    const float* xr = x + (size_t)tok * HH;

    float v[4];
    float s = 0.0f;
#pragma unroll
    for (int i = 0; i < 4; ++i) { v[i] = xr[tid + i * 256]; s += v[i]; }

    __shared__ float red[256];
    __shared__ float sbuf[EE];

    red[tid] = s; __syncthreads();
    for (int off = 128; off > 0; off >>= 1) {
        if (tid < off) red[tid] += red[tid + off];
        __syncthreads();
    }
    const float mu = red[0] * (1.0f / HH);
    __syncthreads();

    float s2 = 0.0f;
#pragma unroll
    for (int i = 0; i < 4; ++i) { float c = v[i] - mu; s2 += c * c; }
    red[tid] = s2; __syncthreads();
    for (int off = 128; off > 0; off >>= 1) {
        if (tid < off) red[tid] += red[tid + off];
        __syncthreads();
    }
    const float rstd = rsqrtf(red[0] * (1.0f / HH) + 1e-5f);
    __syncthreads();

    float* xo = xhat + (size_t)tok * HH;
#pragma unroll
    for (int i = 0; i < 4; ++i) xo[tid + i * 256] = (v[i] - mu) * rstd;

    // router logits on raw x
    float lg[EE] = {0.f, 0.f, 0.f, 0.f};
#pragma unroll
    for (int i = 0; i < 4; ++i) {
        const float* rwh = rw + (size_t)(tid + i * 256) * EE;
#pragma unroll
        for (int e = 0; e < EE; ++e) lg[e] += v[i] * rwh[e];
    }
    for (int e = 0; e < EE; ++e) {
        red[tid] = lg[e]; __syncthreads();
        for (int off = 128; off > 0; off >>= 1) {
            if (tid < off) red[tid] += red[tid + off];
            __syncthreads();
        }
        if (tid == 0) sbuf[e] = red[0] + rb[e];
        __syncthreads();
    }

    if (tid == 0) {
        float p[EE];
        float mx = sbuf[0];
        for (int e = 1; e < EE; ++e) mx = fmaxf(mx, sbuf[e]);
        float sm = 0.0f;
        for (int e = 0; e < EE; ++e) { p[e] = __expf(sbuf[e] - mx); sm += p[e]; }
        for (int e = 0; e < EE; ++e) p[e] /= sm;
        int i0 = 0;
        for (int e = 1; e < EE; ++e) if (p[e] > p[i0]) i0 = e;       // first max
        int i1 = -1;
        for (int e = 0; e < EE; ++e) {
            if (e == i0) continue;
            if (i1 < 0 || p[e] > p[i1]) i1 = e;                       // second max
        }
        const float t = p[i0] + p[i1];
        float* wr = wt + (size_t)tok * EE;
        for (int e = 0; e < EE; ++e) wr[e] = 0.0f;
        wr[i0] = p[i0] / t;
        wr[i1] = p[i1] / t;
    }
}

// ---------------------------------------------------------------------------
// bf16 WMMA GEMM:  C[M,N] = A[M,K] * B[K,N]
//   epi 0 : hOut = bf16(gelu(C + bias))                 (GEMM1)
//   epi 1 : accOut[m,n] += wt[m,expert] * (C + bias[n]) (GEMM2, fused combine)
// Block tile 128x128, BK=32, 256 threads = 8 waves, wave tile 64x32 (4x2 WMMA).
// Double-buffered LDS + register prefetch: one barrier per K step; global-load
// latency for tile k+1 hides behind the 8 WMMAs of tile k.
// ---------------------------------------------------------------------------
#define BM 128
#define BN 128
#define BK 32
#define APAD 8    // A row = 40 bf16 (80 B)
#define BPAD 4    // B row = 132 u32 (528 B)

union AFrag { uint4 u[2]; v16bf v; };
union BFrag { unsigned int w[8]; v16bf v; };

__global__ __launch_bounds__(256) void gemm_bf16(
    const unsigned short* __restrict__ A,   // [M,Kd] bf16 row-major
    const unsigned short* __restrict__ B,   // [Kd,N] bf16 row-major
    const float* __restrict__ bias,         // [N]
    int M, int N, int Kd, int epi,
    unsigned short* __restrict__ hOut,      // epi 0
    float* __restrict__ accOut,             // epi 1
    const float* __restrict__ wt,           // [M,EE]
    int expert) {

    __shared__ __align__(16) unsigned short As[2][BM][BK + APAD];
    __shared__ __align__(16) unsigned int   Bs[2][BK / 2][BN + BPAD];

    const int tid  = threadIdx.x;
    const int lane = tid & 31;
    const int wid  = tid >> 5;
    const int wm   = (wid & 1) * 64;   // wave M offset in block
    const int wn   = (wid >> 1) * 32;  // wave N offset in block
    const int half = lane >> 4;        // K-half selector per ISA layout
    const int l16  = lane & 15;

    const int m0 = blockIdx.y * BM;
    const int n0 = blockIdx.x * BN;

    v8f acc[4][2];
#pragma unroll
    for (int i = 0; i < 4; ++i)
#pragma unroll
        for (int j = 0; j < 2; ++j) acc[i][j] = zero8();

    // A staging: thread -> (row, 16-elem K segment)
    const int ar = tid >> 1;
    const int ac = (tid & 1) * 16;
    // B staging: thread -> (K pair row, 8-col group)
    const int bp = tid >> 4;
    const int bn = (tid & 15) * 8;

    const unsigned short* agBase = A + (size_t)(m0 + ar) * Kd + ac;
    const unsigned short* bgBase = B + (size_t)(2 * bp) * N + n0 + bn;

    uint4 pa0, pa1, pbe, pbo;   // register prefetch staging

    // ---- prologue: fetch tile 0 and commit to LDS buffer 0 ----
    pa0 = *(const uint4*)agBase;
    pa1 = *(const uint4*)(agBase + 8);
    pbe = *(const uint4*)bgBase;
    pbo = *(const uint4*)(bgBase + N);
    {
        *(uint4*)&As[0][ar][ac]     = pa0;
        *(uint4*)&As[0][ar][ac + 8] = pa1;
        unsigned int q0 = (pbe.x & 0xFFFFu) | (pbo.x << 16);
        unsigned int q1 = (pbe.x >> 16)     | (pbo.x & 0xFFFF0000u);
        unsigned int q2 = (pbe.y & 0xFFFFu) | (pbo.y << 16);
        unsigned int q3 = (pbe.y >> 16)     | (pbo.y & 0xFFFF0000u);
        unsigned int q4 = (pbe.z & 0xFFFFu) | (pbo.z << 16);
        unsigned int q5 = (pbe.z >> 16)     | (pbo.z & 0xFFFF0000u);
        unsigned int q6 = (pbe.w & 0xFFFFu) | (pbo.w << 16);
        unsigned int q7 = (pbe.w >> 16)     | (pbo.w & 0xFFFF0000u);
        *(uint4*)&Bs[0][bp][bn]     = make_uint4(q0, q1, q2, q3);
        *(uint4*)&Bs[0][bp][bn + 4] = make_uint4(q4, q5, q6, q7);
    }
    __syncthreads();

    const int nk = Kd / BK;
    for (int ki = 0; ki < nk; ++ki) {
        const int cur = ki & 1;
        const bool more = (ki + 1) < nk;

        // ---- issue next tile's global loads (latency hidden by WMMAs) ----
        if (more) {
            const int k1 = (ki + 1) * BK;
            pa0 = *(const uint4*)(agBase + k1);
            pa1 = *(const uint4*)(agBase + k1 + 8);
            const unsigned short* bg = bgBase + (size_t)k1 * N;
            pbe = *(const uint4*)bg;
            pbo = *(const uint4*)(bg + N);
        }

        // ---- LDS -> fragments (buffer `cur`) ----
        AFrag af[4];
#pragma unroll
        for (int mi = 0; mi < 4; ++mi) {
            const int row = wm + mi * 16 + l16;
            af[mi].u[0] = *(const uint4*)&As[cur][row][half * 8];
            af[mi].u[1] = *(const uint4*)&As[cur][row][16 + half * 8];
        }
        BFrag bf[2];
#pragma unroll
        for (int ni = 0; ni < 2; ++ni) {
            const int col = wn + ni * 16 + l16;
#pragma unroll
            for (int v = 0; v < 8; ++v) bf[ni].w[v] = Bs[cur][half * 4 + v + (v & 4)][col];
        }

        // ---- 8 WMMAs per wave per K step ----
#pragma unroll
        for (int mi = 0; mi < 4; ++mi)
#pragma unroll
            for (int ni = 0; ni < 2; ++ni)
                acc[mi][ni] = __builtin_amdgcn_wmma_f32_16x16x32_bf16(
                    false, af[mi].v, false, bf[ni].v,
                    (short)0, acc[mi][ni], false, false);

        // ---- commit prefetched tile into the other buffer ----
        if (more) {
            const int nxt = cur ^ 1;
            *(uint4*)&As[nxt][ar][ac]     = pa0;
            *(uint4*)&As[nxt][ar][ac + 8] = pa1;
            unsigned int q0 = (pbe.x & 0xFFFFu) | (pbo.x << 16);
            unsigned int q1 = (pbe.x >> 16)     | (pbo.x & 0xFFFF0000u);
            unsigned int q2 = (pbe.y & 0xFFFFu) | (pbo.y << 16);
            unsigned int q3 = (pbe.y >> 16)     | (pbo.y & 0xFFFF0000u);
            unsigned int q4 = (pbe.z & 0xFFFFu) | (pbo.z << 16);
            unsigned int q5 = (pbe.z >> 16)     | (pbo.z & 0xFFFF0000u);
            unsigned int q6 = (pbe.w & 0xFFFFu) | (pbo.w << 16);
            unsigned int q7 = (pbe.w >> 16)     | (pbo.w & 0xFFFF0000u);
            *(uint4*)&Bs[nxt][bp][bn]     = make_uint4(q0, q1, q2, q3);
            *(uint4*)&Bs[nxt][bp][bn + 4] = make_uint4(q4, q5, q6, q7);
        }
        __syncthreads();
    }

    // ---- epilogue (C layout: lanes 0-15 M=r, lanes 16-31 M=r+8; N=l16) ----
#pragma unroll
    for (int mi = 0; mi < 4; ++mi) {
#pragma unroll
        for (int ni = 0; ni < 2; ++ni) {
            const int gn    = n0 + wn + ni * 16 + l16;
            const float bv  = bias[gn];
            const int mbase = m0 + wm + mi * 16 + half * 8;
            if (epi == 0) {
#pragma unroll
                for (int r = 0; r < 8; ++r) {
                    float xv = acc[mi][ni][r] + bv;
                    float gl = 0.5f * xv * (1.0f + erff(xv * 0.70710678118654752f));
                    hOut[(size_t)(mbase + r) * N + gn] = f2bf(gl);
                }
            } else {
#pragma unroll
                for (int r = 0; r < 8; ++r) {
                    const int gm = mbase + r;
                    const float w = wt[(size_t)gm * EE + expert];
                    accOut[(size_t)gm * N + gn] += w * (acc[mi][ni][r] + bv);
                }
            }
        }
    }
}

// ---------------------------------------------------------------------------
// Host orchestration
// ---------------------------------------------------------------------------
extern "C" void kernel_launch(void* const* d_in, const int* in_sizes, int n_in,
                              void* d_out, int out_size, void* d_ws, size_t ws_size,
                              hipStream_t stream) {
    (void)in_sizes; (void)n_in; (void)out_size; (void)ws_size;

    const float* x_in     = (const float*)d_in[0];
    const float* router_w = (const float*)d_in[1];
    const float* router_b = (const float*)d_in[2];
    const float* ln_g     = (const float*)d_in[3];
    const float* ln_b     = (const float*)d_in[4];
    const float* w1       = (const float*)d_in[5];
    const float* b1       = (const float*)d_in[6];
    const float* w2       = (const float*)d_in[7];
    const float* b2       = (const float*)d_in[8];
    float* out = (float*)d_out;

    char* ws = (char*)d_ws;
    size_t off = 0;
    auto salloc = [&](size_t bytes) -> char* {
        char* p = ws + off;
        off += (bytes + 255) & ~(size_t)255;
        return p;
    };
    unsigned short* wbuf1 = (unsigned short*)salloc((size_t)HH * FF * 2);   // 8.4 MB
    unsigned short* wbuf2 = (unsigned short*)salloc((size_t)FF * HH * 2);   // 8.4 MB
    float*          xhat  = (float*)salloc((size_t)NTOK * HH * 4);          // 33.5 MB
    float*          wt    = (float*)salloc((size_t)NTOK * EE * 4);          // 128 KB
    unsigned short* xe    = (unsigned short*)salloc((size_t)NTOK * HH * 2); // 16.8 MB
    unsigned short* hbuf  = (unsigned short*)salloc((size_t)NTOK * FF * 2); // 67 MB
    float*          xping = (float*)salloc((size_t)NTOK * HH * 4);          // 33.5 MB

    const int cvtThreads = 256;
    const int nW4  = HH * FF / 4;          // per-expert weight float4 count
    const int nX4  = NTOK * HH / 4;        // activation float4 count

    const float* xcur = x_in;
    for (int l = 0; l < LL; ++l) {
        // acc target: L0->xping, L1->out, L2->xping, L3->out (final in d_out)
        float* accT = (l & 1) ? out : xping;

        router_ln<<<NTOK, 256, 0, stream>>>(
            xcur, router_w + (size_t)l * HH * EE, router_b + (size_t)l * EE,
            xhat, wt);
        copy_f32<<<(nX4 + cvtThreads - 1) / cvtThreads, cvtThreads, 0, stream>>>(
            xcur, accT, nX4);

        for (int e = 0; e < EE; ++e) {
            const size_t wi = (size_t)l * EE + e;
            cvt_f32_bf16<<<(nW4 + cvtThreads - 1) / cvtThreads, cvtThreads, 0, stream>>>(
                w1 + wi * (size_t)HH * FF, wbuf1, nW4);
            cvt_f32_bf16<<<(nW4 + cvtThreads - 1) / cvtThreads, cvtThreads, 0, stream>>>(
                w2 + wi * (size_t)FF * HH, wbuf2, nW4);
            apply_gb<<<(nX4 + cvtThreads - 1) / cvtThreads, cvtThreads, 0, stream>>>(
                xhat, ln_g + wi * HH, ln_b + wi * HH, xe, nX4);

            dim3 g1(FF / BN, NTOK / BM);   // (32, 64)
            gemm_bf16<<<g1, 256, 0, stream>>>(
                xe, wbuf1, b1 + wi * FF, NTOK, FF, HH, /*epi=*/0,
                hbuf, nullptr, nullptr, 0);

            dim3 g2(HH / BN, NTOK / BM);   // (8, 64)
            gemm_bf16<<<g2, 256, 0, stream>>>(
                hbuf, wbuf2, b2 + wi * HH, NTOK, HH, FF, /*epi=*/1,
                nullptr, accT, wt, e);
        }
        xcur = accT;
    }
}
